// GCN_extract_part_18176301596817
// MI455X (gfx1250) — matrix operand order repaired and checked by reading, not compiled, not run on hardware
//
#include <hip/hip_runtime.h>

// ---------------------------------------------------------------------------
// Types for CDNA5 WMMA (wave32)
// ---------------------------------------------------------------------------
typedef __attribute__((ext_vector_type(16))) __bf16 v16bf;
typedef __attribute__((ext_vector_type(8)))  __bf16 v8bf;
typedef __attribute__((ext_vector_type(4)))  __bf16 v4bf;
typedef __attribute__((ext_vector_type(8)))  float  v8f;

__device__ __forceinline__ __bf16 f2bf(float f) {
    unsigned u = __builtin_bit_cast(unsigned, f);
    unsigned r = u + 0x7FFFu + ((u >> 16) & 1u);   // round-to-nearest-even
    unsigned short hh = (unsigned short)(r >> 16);
    return __builtin_bit_cast(__bf16, hh);
}

// ---------------------------------------------------------------------------
// Utility fills / conversions
// ---------------------------------------------------------------------------
__global__ __launch_bounds__(256) void fill_zero_u4(uint4* __restrict__ p, long n16) {
    long i = (long)blockIdx.x * blockDim.x + threadIdx.x;
    if (i < n16) p[i] = make_uint4(0u, 0u, 0u, 0u);
}

// pad X [rows,K] f32 -> [rows,Kpad] bf16 (zero pad)
__global__ __launch_bounds__(256) void pad_x_bf16(const float* __restrict__ X,
                                                  __bf16* __restrict__ Xb,
                                                  long n, int K, int Kpad) {
    long i = (long)blockIdx.x * blockDim.x + threadIdx.x;
    if (i >= n) return;
    long r = i / Kpad;
    int  k = (int)(i - r * (long)Kpad);
    Xb[i] = (k < K) ? f2bf(X[r * (long)K + k]) : f2bf(0.0f);
}

// W [K,256] f32 -> Wt [256,Kpad] bf16 (transposed, zero K-pad)
__global__ __launch_bounds__(256) void wt_bf16_kernel(const float* __restrict__ W,
                                                      __bf16* __restrict__ Wt,
                                                      int K, int Kpad) {
    long i = (long)blockIdx.x * blockDim.x + threadIdx.x;
    if (i >= 256L * Kpad) return;
    int n = (int)(i / Kpad);
    int k = (int)(i - (long)n * Kpad);
    Wt[i] = (k < K) ? f2bf(W[(size_t)k * 256 + n]) : f2bf(0.0f);
}

// ---------------------------------------------------------------------------
// Degree / normalization
// ---------------------------------------------------------------------------
__global__ __launch_bounds__(256) void degree_kernel(const int* __restrict__ dst, int E,
                                                     int* __restrict__ deg) {
    int e = blockIdx.x * blockDim.x + threadIdx.x;
    if (e < E) atomicAdd(&deg[dst[e]], 1);
}

__global__ __launch_bounds__(256) void deg_inv_kernel(const int* __restrict__ deg,
                                                      float* __restrict__ dinv,
                                                      float* __restrict__ invdeg, int n) {
    int i = blockIdx.x * blockDim.x + threadIdx.x;
    if (i >= n) return;
    float d = (float)(deg[i] + 1);          // +1 self loop
    dinv[i]   = rsqrtf(d);
    invdeg[i] = 1.0f / d;
}

// ---------------------------------------------------------------------------
// bf16 WMMA GEMM:  C[M,256] = A[M,Kpad] * Bt[256,Kpad]^T   (f32 accumulate)
// block = 256 threads (8 waves), block tile 128x128, wave tile 16x128.
// A and B fragments use the same contiguous k order: element s of lane half h = k0 + 16h + s.
// ---------------------------------------------------------------------------
__global__ __launch_bounds__(256) void gemm_bf16_wmma(const __bf16* __restrict__ A,
                                                      const __bf16* __restrict__ Bt,
                                                      float* __restrict__ C, int Kpad) {
    const int lane  = threadIdx.x & 31;
    const int wave  = threadIdx.x >> 5;
    const int half  = lane >> 4;     // 0: lanes 0-15, 1: lanes 16-31
    const int lidx  = lane & 15;
    const int mBase = blockIdx.x * 128 + wave * 16;
    const int nBase = blockIdx.y * 128;

    v8f acc[8] = {};
    const __bf16* aRow = A + (size_t)(mBase + lidx) * Kpad;

    for (int k0 = 0; k0 < Kpad; k0 += 32) {
        // A 16x32 bf16: elems 0..15 = K(k0+half*16 ..)
        v8bf alo = *(const v8bf*)(aRow + k0 + half * 16);
        v8bf ahi = *(const v8bf*)(aRow + k0 + half * 16 + 8);
        v16bf afrag = __builtin_shufflevector(alo, ahi,
            0, 1, 2, 3, 4, 5, 6, 7, 8, 9, 10, 11, 12, 13, 14, 15);
#pragma unroll
        for (int t = 0; t < 8; ++t) {
            // B 32x16 bf16: lane col = nT+lidx; elems 0..15 = K(k0+half*16 ..)
            const __bf16* bp = Bt + (size_t)(nBase + t * 16 + lidx) * Kpad + k0 + half * 16;
            v8bf b0 = *(const v8bf*)(bp);
            v8bf b1 = *(const v8bf*)(bp + 8);
            v16bf bfrag = __builtin_shufflevector(b0, b1,
                0, 1, 2, 3, 4, 5, 6, 7, 8, 9, 10, 11, 12, 13, 14, 15);
            acc[t] = __builtin_amdgcn_wmma_f32_16x16x32_bf16(
                false, afrag, false, bfrag, (short)0, acc[t], false, false);
        }
    }

    // C 16x16 f32 layout: VGPR r -> row mBase + half*8 + r, col = nT + lidx
    const int row0 = mBase + half * 8;
#pragma unroll
    for (int t = 0; t < 8; ++t) {
        const int col = nBase + t * 16 + lidx;
#pragma unroll
        for (int r = 0; r < 8; ++r)
            C[(size_t)(row0 + r) * 256 + col] = acc[t][r];
    }
}

// ---------------------------------------------------------------------------
// Edge aggregation: agg[dst] += h[src] * dinv[src]*dinv[dst]  (f32 atomics, L2)
// one wave per edge; lane handles float4 at positions lane and lane+32
// ---------------------------------------------------------------------------
__global__ __launch_bounds__(256) void edge_agg_kernel(const int* __restrict__ src,
                                                       const int* __restrict__ dst,
                                                       const float* __restrict__ h,
                                                       const float* __restrict__ dinv,
                                                       float* __restrict__ agg, int E) {
    int e = blockIdx.x * 8 + (threadIdx.x >> 5);
    if (e >= E) return;
    int lane = threadIdx.x & 31;
    int s = src[e], d = dst[e];
    float norm = dinv[s] * dinv[d];
    const float4* hp = (const float4*)(h + (size_t)s * 256);
    float* ap = agg + (size_t)d * 256;
#pragma unroll
    for (int i = 0; i < 2; ++i) {
        int c4 = lane + i * 32;
        float4 v = hp[c4];
        int c = c4 * 4;
        atomicAdd(ap + c + 0, v.x * norm);
        atomicAdd(ap + c + 1, v.y * norm);
        atomicAdd(ap + c + 2, v.z * norm);
        atomicAdd(ap + c + 3, v.w * norm);
    }
}

// ---------------------------------------------------------------------------
// combine: out = relu(agg + h*invdeg + bias)  -> bf16 (layers 1,2)
// ---------------------------------------------------------------------------
__global__ __launch_bounds__(256) void combine_relu_kernel(const float4* __restrict__ agg,
                                                           const float4* __restrict__ h,
                                                           const float* __restrict__ invdeg,
                                                           const float4* __restrict__ bias,
                                                           v4bf* __restrict__ out, long n4) {
    long i = (long)blockIdx.x * blockDim.x + threadIdx.x;
    if (i >= n4) return;
    int row = (int)(i >> 6);                 // 64 float4 per 256-wide row
    float id = invdeg[row];
    float4 a = agg[i], hh = h[i], b = bias[i & 63];
    v4bf o;
    o[0] = f2bf(fmaxf(a.x + hh.x * id + b.x, 0.0f));
    o[1] = f2bf(fmaxf(a.y + hh.y * id + b.y, 0.0f));
    o[2] = f2bf(fmaxf(a.z + hh.z * id + b.z, 0.0f));
    o[3] = f2bf(fmaxf(a.w + hh.w * id + b.w, 0.0f));
    out[i] = o;
}

// final layer: agg <- agg + h*invdeg + bias  (f32 in-place, no relu)
__global__ __launch_bounds__(256) void combine_final_kernel(float4* __restrict__ agg,
                                                            const float4* __restrict__ h,
                                                            const float* __restrict__ invdeg,
                                                            const float4* __restrict__ bias,
                                                            long n4) {
    long i = (long)blockIdx.x * blockDim.x + threadIdx.x;
    if (i >= n4) return;
    int row = (int)(i >> 6);
    float id = invdeg[row];
    float4 a = agg[i], hh = h[i], b = bias[i & 63];
    a.x += hh.x * id + b.x;
    a.y += hh.y * id + b.y;
    a.z += hh.z * id + b.z;
    a.w += hh.w * id + b.w;
    agg[i] = a;
}

// ---------------------------------------------------------------------------
// Pack-to-padded
// ---------------------------------------------------------------------------
__global__ __launch_bounds__(256) void starts_kernel(const int* __restrict__ lens,
                                                     int* __restrict__ starts, int Bn) {
    int g = blockIdx.x * blockDim.x + threadIdx.x;
    if (g >= Bn) return;
    int s = 0;
    for (int i = 0; i < g; ++i) s += lens[i];   // B=2048, L2-resident, cheap
    starts[g] = s;
}

__global__ __launch_bounds__(256) void pack_kernel(const float* __restrict__ nodes,
                                                   const int* __restrict__ lens,
                                                   const int* __restrict__ starts,
                                                   float* __restrict__ out, int maxa) {
    int g = blockIdx.x;
    int len = lens[g];
    if (len > maxa) len = maxa;
    int start = starts[g];
    int c = threadIdx.x;                        // 256 features
    for (int p = 0; p < len; ++p)
        out[((size_t)g * maxa + p) * 256 + c] = nodes[(size_t)(start + p) * 256 + c];
}

// ---------------------------------------------------------------------------
// Launch
// ---------------------------------------------------------------------------
static inline long cdiv(long a, long b) { return (a + b - 1) / b; }

extern "C" void kernel_launch(void* const* d_in, const int* in_sizes, int n_in,
                              void* d_out, int out_size, void* d_ws, size_t ws_size,
                              hipStream_t stream) {
    const int D_IN = 75, MAXA = 100;
    const float* drug_x   = (const float*)d_in[0];
    const int*   drug_ei  = (const int*)d_in[1];
    const int*   drug_len = (const int*)d_in[2];
    const float* food_x   = (const float*)d_in[3];
    const int*   food_ei  = (const int*)d_in[4];
    const int*   food_len = (const int*)d_in[5];
    const float* dW[3] = {(const float*)d_in[6],  (const float*)d_in[8],  (const float*)d_in[10]};
    const float* dB[3] = {(const float*)d_in[7],  (const float*)d_in[9],  (const float*)d_in[11]};
    const float* fW[3] = {(const float*)d_in[12], (const float*)d_in[14], (const float*)d_in[16]};
    const float* fB[3] = {(const float*)d_in[13], (const float*)d_in[15], (const float*)d_in[17]};

    const int N  = in_sizes[0] / D_IN;   // 131072
    const int E  = in_sizes[1] / 2;      // 524288
    const int Bn = in_sizes[2];          // 2048

    // ---- workspace carve (256B aligned) ----
    size_t off = 0;
    auto carve = [&](size_t bytes) -> void* {
        void* p = (char*)d_ws + off;
        off += (bytes + 255) & ~(size_t)255;
        return p;
    };
    __bf16* xb     = (__bf16*)carve((size_t)N * 96 * 2);    // padded bf16 input
    __bf16* wt     = (__bf16*)carve((size_t)256 * 256 * 2); // bf16 transposed weights
    float*  h      = (float*)carve((size_t)N * 256 * 4);    // x @ W
    float*  agg    = (float*)carve((size_t)N * 256 * 4);    // scatter accumulator
    __bf16* hb     = (__bf16*)carve((size_t)N * 256 * 2);   // bf16 activations
    int*    deg    = (int*)carve((size_t)N * 4);
    float*  dinv   = (float*)carve((size_t)N * 4);
    float*  invdeg = (float*)carve((size_t)N * 4);
    int*    starts = (int*)carve((size_t)Bn * 4);

    // zero entire output (pack writes only valid rows)
    {
        long n16 = (long)out_size * 4 / 16;
        fill_zero_u4<<<cdiv(n16, 256), 256, 0, stream>>>((uint4*)d_out, n16);
    }

    float* outs[2] = {(float*)d_out, (float*)d_out + (size_t)Bn * MAXA * 256};

    for (int side = 0; side < 2; ++side) {
        const float* X    = side ? food_x   : drug_x;
        const int*   srcp = side ? food_ei  : drug_ei;
        const int*   dstp = srcp + E;
        const int*   lens = side ? food_len : drug_len;
        const float* const* W    = side ? fW : dW;
        const float* const* Bias = side ? fB : dB;

        // degrees + normalization terms
        fill_zero_u4<<<cdiv((long)N / 4, 256), 256, 0, stream>>>((uint4*)deg, (long)N / 4);
        degree_kernel<<<cdiv(E, 256), 256, 0, stream>>>(dstp, E, deg);
        deg_inv_kernel<<<cdiv(N, 256), 256, 0, stream>>>(deg, dinv, invdeg, N);

        // pad/convert input features
        {
            long n = (long)N * 96;
            pad_x_bf16<<<cdiv(n, 256), 256, 0, stream>>>(X, xb, n, D_IN, 96);
        }

        for (int layer = 0; layer < 3; ++layer) {
            const int K    = (layer == 0) ? D_IN : 256;
            const int Kpad = (layer == 0) ? 96   : 256;
            const __bf16* Ain = (layer == 0) ? xb : hb;

            // transpose+convert weights
            wt_bf16_kernel<<<cdiv(256L * Kpad, 256), 256, 0, stream>>>(W[layer], wt, K, Kpad);

            // h = A @ W   (bf16 WMMA, f32 accumulate)
            dim3 grid(N / 128, 2);
            gemm_bf16_wmma<<<grid, 256, 0, stream>>>(Ain, wt, h, Kpad);

            // agg = scatter_add(h[src] * norm)
            fill_zero_u4<<<cdiv((long)N * 64, 256), 256, 0, stream>>>((uint4*)agg, (long)N * 64);
            edge_agg_kernel<<<cdiv(E, 8), 256, 0, stream>>>(srcp, dstp, h, dinv, agg, E);

            // out = agg + h/deg + b  (+relu for layers 0,1)
            long n4 = (long)N * 64;
            if (layer < 2)
                combine_relu_kernel<<<cdiv(n4, 256), 256, 0, stream>>>(
                    (const float4*)agg, (const float4*)h, invdeg,
                    (const float4*)Bias[layer], (v4bf*)hb, n4);
            else
                combine_final_kernel<<<cdiv(n4, 256), 256, 0, stream>>>(
                    (float4*)agg, (const float4*)h, invdeg,
                    (const float4*)Bias[layer], n4);
        }

        // pack to [B, 100, 256]
        starts_kernel<<<cdiv(Bn, 256), 256, 0, stream>>>(lens, starts, Bn);
        pack_kernel<<<Bn, 256, 0, stream>>>(agg, lens, starts, outs[side], MAXA);
    }
}
